// QKV_Attention_3341484557034
// MI455X (gfx1250) — matrix operand order repaired and checked by reading, not compiled, not run on hardware
//
#include <hip/hip_runtime.h>
#include <hip/hip_bf16.h>

typedef __attribute__((ext_vector_type(16))) _Float16 v16h;
typedef __attribute__((ext_vector_type(8)))  float    v8f;
typedef __attribute__((ext_vector_type(4)))  unsigned int v4u;
typedef __attribute__((ext_vector_type(8)))  int      v8i;
typedef __attribute__((ext_vector_type(4)))  int      v4i;

constexpr int BB = 4;      // batch
constexpr int LL = 1024;   // sequence
constexpr int EE = 512;    // embed
constexpr int KH = 8;      // heads
constexpr int KE = KH * EE;          // 4096
constexpr float SCALE = 0.04419417382415922f; // 1/sqrt(512)

// ---------------- fragment loaders (ISA 7.12.2 layouts) ----------------
// A-matrix 16x32 f16: lane = hi*16+idx holds row (rowbase+idx);
// halves 0..7  = K in [kb + hi*8,      +8)
// halves 8..15 = K in [kb + 16 + hi*8, +8)
__device__ __forceinline__ v16h loadA(const _Float16* src, int stride,
                                      int rowbase, int kb, int idx, int hi) {
    union { v16h v; uint4 q[2]; } f;
    const _Float16* p = src + (rowbase + idx) * stride + kb;
    f.q[0] = *(const uint4*)(p + hi * 8);
    f.q[1] = *(const uint4*)(p + 16 + hi * 8);
    return f.v;
}

// B-matrix 32x16 f16: lane = hi*16+idx holds column (colbase+idx);
// halves 0..7 = K in [kb + hi*8, +8); halves 8..15 = K in [kb + 16 + hi*8, +8).
// srcT is stored so that "column c" of B is a contiguous row of srcT.
__device__ __forceinline__ v16h loadB(const _Float16* srcT, int stride,
                                      int colbase, int kb, int idx, int hi) {
    union { v16h v; uint4 q[2]; } f;
    const _Float16* p = srcT + (colbase + idx) * stride + kb;
    f.q[0] = *(const uint4*)(p + hi * 8);
    f.q[1] = *(const uint4*)(p + 16 + hi * 8);
    return f.v;
}

__device__ __forceinline__ v8f wmma32(v16h a, v16h b, v8f c) {
    return __builtin_amdgcn_wmma_f32_16x16x32_f16(
        false, a, false, b, (short)0, c, false, false);
}

// ---------------- kernel 1: fp32 -> f16 conversions ----------------
__global__ void convert_kernel(const float* __restrict__ xsa,
                               const float* __restrict__ Wq,
                               _Float16* __restrict__ Xh,    // (B*L, E) row-major
                               _Float16* __restrict__ Wqh,   // (KE, E) row-major
                               _Float16* __restrict__ Xt) {  // per batch (E, L)
    const int nX = BB * LL * EE;
    const int nW = KE * EE;
    int stride = gridDim.x * blockDim.x;
    for (int i = blockIdx.x * blockDim.x + threadIdx.x; i < nX; i += stride) {
        float v = xsa[i];
        Xh[i] = (_Float16)v;
        int b = i / (LL * EE);
        int r = i - b * (LL * EE);
        int l = r / EE;
        int e = r - l * EE;
        Xt[(b * EE + e) * LL + l] = (_Float16)v;
    }
    for (int i = blockIdx.x * blockDim.x + threadIdx.x; i < nW; i += stride) {
        Wqh[i] = (_Float16)Wq[i];
    }
}

// ---------------- kernel 2: Q = X * Wq^T  (f16 in, f16 out, f32 acc) ----------------
// M = B*L = 4096 rows, N = KE = 4096 cols. One 16x16 tile per wave.
__global__ void qgemm_kernel(const _Float16* __restrict__ Xh,
                             const _Float16* __restrict__ Wqh,
                             _Float16* __restrict__ Qh) {   // (B*L, KE)
    const int lane = threadIdx.x & 31;
    const int idx  = lane & 15;
    const int hi   = lane >> 4;
    const int tile = blockIdx.x * (blockDim.x >> 5) + (threadIdx.x >> 5);
    const int MT = (BB * LL) / 16;   // 256
    const int NT = KE / 16;          // 256
    if (tile >= MT * NT) return;
    const int tm = tile >> 8;        // / NT
    const int tn = tile & 255;       // % NT

    v8f acc = {};
    for (int ck = 0; ck < EE / 32; ++ck) {
        __builtin_prefetch(Xh + (tm * 16 + idx) * EE + ck * 32 + 64, 0, 1);
        v16h a = loadA(Xh,  EE, tm * 16, ck * 32, idx, hi);
        v16h b = loadB(Wqh, EE, tn * 16, ck * 32, idx, hi); // B col o = Wq row o (over e)
        acc = wmma32(a, b, acc);
    }
    // D layout: vgpr r -> row r + 8*hi, col = idx
    for (int r = 0; r < 8; ++r) {
        int row = tm * 16 + r + 8 * hi;
        int col = tn * 16 + idx;
        Qh[row * KE + col] = (_Float16)acc[r];
    }
}

// ---------------- kernel 3: fused attention per (b, head, 16-row tile) --------------
__global__ void attn_kernel(const _Float16* __restrict__ Qh,  // (B*L, KE) f16
                            const _Float16* __restrict__ Xh,  // (B*L, E)  f16 (K and V)
                            const _Float16* __restrict__ Xt,  // (B, E, L) f16
                            float* __restrict__ xid,          // (B, L, KH, E)
                            float* __restrict__ pijk) {       // (B, L, KH, L)
    __shared__ _Float16 qtile[16 * EE];   // 16 KB
    __shared__ _Float16 pbuf[16 * LL];    // 32 KB
    __shared__ float red[16][16];
    __shared__ float rowmax[16], rowsum[16];

    const int bid = blockIdx.x;                 // [0, B*KH*(L/16))
    const int b  = bid / (KH * (LL / 16));
    const int r0 = bid - b * (KH * (LL / 16));
    const int k  = r0 / (LL / 16);
    const int lt = r0 - k * (LL / 16);

    const int tid  = threadIdx.x;               // 256 threads = 8 waves
    const int wave = tid >> 5;
    const int lane = tid & 31;
    const int idx  = lane & 15;
    const int hi   = lane >> 4;

    // ---- stage Q tile (16 rows x E f16) into LDS via the Tensor Data Mover ----
    // 2D tile: tile_dim0 = 512 elements (one head's E slice), tile_dim1 = 16 rows,
    // row stride = KE elements, data_size = 2 bytes. Issued once by wave 0.
    if (tid < 32) {
        const unsigned long long gaddr =
            (unsigned long long)(uintptr_t)(Qh + (size_t)(b * LL + lt * 16) * KE + k * EE);
        const unsigned int lds_off = (unsigned int)(uintptr_t)(void*)qtile; // addr[31:0] = LDS offset

        // D# group 0: [1:0]=count=1 | [63:32]=lds_addr | [120:64]=global_addr | [127:126]=type 2
        v4u g0;
        g0[0] = 1u;
        g0[1] = lds_off;
        g0[2] = (unsigned int)gaddr;
        g0[3] = ((unsigned int)(gaddr >> 32) & 0x01FFFFFFu) | (2u << 30);

        // D# group 1: wg_mask=0, data_size=1(2B), no barrier/iterate/pad;
        // tensor_dim0 = KE, tensor_dim1 = B*L, tile_dim0 = EE, tile_dim1 = 16,
        // tensor_dim0_stride = KE (elements), tensor_dim1_stride = 0.
        v8i g1;
        g1[0] = (int)(1u << 16);                                  // data_size = 2 bytes
        g1[1] = (int)(((unsigned)KE & 0xFFFFu) << 16);            // tensor_dim0[15:0] -> [63:48]
        g1[2] = (int)((((unsigned)KE >> 16) & 0xFFFFu)            // tensor_dim0[31:16]
                      | (((unsigned)(BB * LL) & 0xFFFFu) << 16)); // tensor_dim1[15:0]
        g1[3] = (int)((((unsigned)(BB * LL) >> 16) & 0xFFFFu)     // tensor_dim1[31:16]
                      | (((unsigned)EE) << 16));                  // tile_dim0 = 512
        g1[4] = 16;                                               // tile_dim1 = 16 rows
        g1[5] = KE;                                               // tensor_dim0_stride[31:0]
        g1[6] = 0;
        g1[7] = 0;

        v4i g2 = {0, 0, 0, 0};   // 2-D tile: dims 2..4 unused
        v4i g3 = {0, 0, 0, 0};
        v8i g4 = {0, 0, 0, 0, 0, 0, 0, 0};  // extra descriptor group (6-arg toolchain form)

        __builtin_amdgcn_tensor_load_to_lds(g0, g1, g2, g3, g4, 0);
        __builtin_amdgcn_s_wait_tensorcnt(0);
    }
    __syncthreads();

    const _Float16* Xb  = Xh + (size_t)b * LL * EE;
    const _Float16* Xtb = Xt + (size_t)b * EE * LL;
    float* prow0 = pijk + ((size_t)(b * LL + lt * 16) * KH + k) * LL;
    const size_t prstride = (size_t)KH * LL;    // row-to-row stride inside pijk

    // ---- phase A: scores S = Q * X^T, wave owns 128 columns (8 col-tiles) ----
    {
        v8f acc[8];
        for (int ct = 0; ct < 8; ++ct) acc[ct] = (v8f){};
        for (int ck = 0; ck < EE / 32; ++ck) {
            v16h a = loadA(qtile, EE, 0, ck * 32, idx, hi);
            for (int ct = 0; ct < 8; ++ct) {
                int colb = wave * 128 + ct * 16;
                v16h bf = loadB(Xb, EE, colb, ck * 32, idx, hi); // B col n = key row n
                acc[ct] = wmma32(a, bf, acc[ct]);
            }
        }
        for (int ct = 0; ct < 8; ++ct) {
            int colb = wave * 128 + ct * 16;
            for (int r = 0; r < 8; ++r) {
                int row = r + 8 * hi;
                prow0[row * prstride + colb + idx] = acc[ct][r] * SCALE;
            }
        }
    }
    __threadfence();
    __syncthreads();

    // ---- phase B: softmax over the 1024 columns of each of the 16 rows ----
    {
        const int row = tid >> 4;       // 0..15
        const int sub = tid & 15;       // 0..15, strided column scan
        float* pr = prow0 + row * prstride;

        float mx = -3.4e38f;
        for (int j = 0; j < 64; ++j) mx = fmaxf(mx, pr[sub + 16 * j]);
        red[row][sub] = mx;
        __syncthreads();
        if (sub == 0) {
            float m2 = red[row][0];
            for (int t = 1; t < 16; ++t) m2 = fmaxf(m2, red[row][t]);
            rowmax[row] = m2;
        }
        __syncthreads();
        float m = rowmax[row];
        float s = 0.f;
        for (int j = 0; j < 64; ++j) s += __expf(pr[sub + 16 * j] - m);
        red[row][sub] = s;
        __syncthreads();
        if (sub == 0) {
            float s2 = 0.f;
            for (int t = 0; t < 16; ++t) s2 += red[row][t];
            rowsum[row] = s2;
        }
        __syncthreads();
        float inv = 1.0f / rowsum[row];
        for (int j = 0; j < 64; ++j) {
            int c = sub + 16 * j;
            float p = __expf(pr[c] - m) * inv;
            pr[c] = p;                               // final pijk output
            pbuf[row * LL + c] = (_Float16)p;        // f16 copy for the V matmul
        }
    }
    __syncthreads();

    // ---- phase C: Y = P * X, wave owns 4 output col-tiles (64 of 512 cols) ----
    {
        v8f acc[4];
        for (int nt = 0; nt < 4; ++nt) acc[nt] = (v8f){};
        for (int ck = 0; ck < LL / 32; ++ck) {
            v16h a = loadA(pbuf, LL, 0, ck * 32, idx, hi);
            for (int nt = 0; nt < 4; ++nt) {
                int eb = (wave * 4 + nt) * 16;
                v16h bf = loadB(Xtb, LL, eb, ck * 32, idx, hi); // B col e = Xt row e (over l)
                acc[nt] = wmma32(a, bf, acc[nt]);
            }
        }
        float* yrow0 = xid + ((size_t)(b * LL + lt * 16) * KH + k) * EE;
        const size_t ystride = (size_t)KH * EE;
        for (int nt = 0; nt < 4; ++nt) {
            int eb = (wave * 4 + nt) * 16;
            for (int r = 0; r < 8; ++r) {
                int row = r + 8 * hi;
                yrow0[row * ystride + eb + idx] = acc[nt][r];
            }
        }
    }
}

extern "C" void kernel_launch(void* const* d_in, const int* in_sizes, int n_in,
                              void* d_out, int out_size, void* d_ws, size_t ws_size,
                              hipStream_t stream) {
    const float* xsa = (const float*)d_in[0];   // (4,1024,512)
    const float* Wq  = (const float*)d_in[1];   // (4096,512)

    // workspace layout (f16): Xh | Wqh | Xt | Qh  = 4+4+4+32 MiB
    _Float16* Xh  = (_Float16*)d_ws;
    _Float16* Wqh = Xh  + (size_t)BB * LL * EE;
    _Float16* Xt  = Wqh + (size_t)KE * EE;
    _Float16* Qh  = Xt  + (size_t)BB * LL * EE;

    float* xid  = (float*)d_out;                              // (4,1024,8,512)
    float* pijk = xid + (size_t)BB * LL * KH * EE;            // (4,1024,8,1024)

    convert_kernel<<<2048, 256, 0, stream>>>(xsa, Wq, Xh, Wqh, Xt);

    // 256x256 output tiles, 8 waves (tiles) per 256-thread block
    qgemm_kernel<<<(256 * 256) / 8, 256, 0, stream>>>(Xh, Wqh, Qh);

    // one block per (b, head, 16-row tile): 4*8*64 = 2048
    attn_kernel<<<BB * KH * (LL / 16), 256, 0, stream>>>(Qh, Xh, Xt, xid, pijk);
}
